// USA_89910845374897
// MI455X (gfx1250) — matrix operand order repaired and checked by reading, not compiled, not run on hardware
//
#include <hip/hip_runtime.h>
#include <hip/hip_bf16.h>
#include <math.h>

// ---------------------------------------------------------------------------
// Types for CDNA5 WMMA (wave32): v_wmma_f32_16x16x32_f16
//   A: 16x32 f16  (v16h, 8 VGPRs)   lane r<16 holds row r, K={0..7,16..23};
//                                    lane 16+r holds row r, K={8..15,24..31}
//   B: 32x16 f16  (v16h, 8 VGPRs)   lane n<16 holds col n, K={0..7,16..23};
//                                    lane 16+n holds col n, K={8..15,24..31}
//   C/D: 16x16 f32 (v8f)            VGPR v, lane n<16 -> D[v][n];
//                                    lane 16+n -> D[8+v][n]
// ---------------------------------------------------------------------------
typedef _Float16 h16 __attribute__((ext_vector_type(16)));
typedef _Float16 h8  __attribute__((ext_vector_type(8)));
typedef float    f8  __attribute__((ext_vector_type(8)));
typedef float    f4v __attribute__((ext_vector_type(4)));

#define NHEADS 32
#define SEQ    2048
#define DIM    128
#define IDM    128
#define FD     32

__device__ __forceinline__ h16 cat8(h8 lo, h8 hi) {
    return __builtin_shufflevector(lo, hi, 0,1,2,3,4,5,6,7,8,9,10,11,12,13,14,15);
}

__device__ __forceinline__ f8 wmma_f16(h16 a, h16 b, f8 c) {
    return __builtin_amdgcn_wmma_f32_16x16x32_f16(
        /*neg_a=*/false, a, /*neg_b=*/false, b,
        /*c_mod=*/(short)0, c, /*reuse_a=*/false, /*reuse_b=*/false);
}

// ---------------------------------------------------------------------------
// Prep: W[a][K][N] fp32  ->  Wt[a][N][K] f16   (B-fragment friendly layout)
// ---------------------------------------------------------------------------
__global__ void USA_transpose_f32_to_h16(const float* __restrict__ in,
                                         _Float16* __restrict__ out,
                                         int K, int N, long total) {
    long i = (long)blockIdx.x * blockDim.x + threadIdx.x;
    if (i >= total) return;
    long kn = (long)K * N;
    long a  = i / kn;
    long r  = i - a * kn;
    int  k  = (int)(r / N);
    int  n  = (int)(r - (long)k * N);
    out[a * kn + (long)n * K + k] = (_Float16)in[i];
}

// ---------------------------------------------------------------------------
// Stage 1: per-head token MLP with tanh.  X[a][s][128] fp32 ->
//          Y[a][s][32] f16 = tanh(silu(silu(X W1 + b1) W2 + b2) W3 + b3)
// Block = 256 threads (8 waves). Wave owns 16 tokens. grid = (SEQ/128, NHEADS)
// ---------------------------------------------------------------------------
__global__ __launch_bounds__(256) void USA_uplift_kernel(
    const float*    __restrict__ X,    // [A, SEQ, DIM] fp32
    const _Float16* __restrict__ W1t,  // [A, IDM, DIM]  (n-major)
    const float*    __restrict__ b1,   // [A, IDM]
    const _Float16* __restrict__ W2t,  // [A, IDM, IDM]
    const float*    __restrict__ b2,   // [A, IDM]
    const _Float16* __restrict__ W3t,  // [A, FD, IDM]
    const float*    __restrict__ b3,   // [A, FD]
    _Float16*       __restrict__ Y)    // [A, SEQ, FD]  f16 tanh output
{
    __shared__ _Float16 Hbuf[8][16 * IDM];   // 32 KB, one 16x128 slab per wave

    const int a    = blockIdx.y;
    const int wave = threadIdx.x >> 5;
    const int lane = threadIdx.x & 31;
    const int lrow = lane & 15;
    const bool hi  = lane >= 16;
    const int m0   = blockIdx.x * 128 + wave * 16;

    _Float16* Hl = &Hbuf[wave][0];

    // ---- A fragments of X (fp32 -> f16), 4 K-steps of 32 ----
    const float* Xrow = X + ((size_t)a * SEQ + m0 + lrow) * DIM;
    h16 ax[4];
    #pragma unroll
    for (int kk = 0; kk < 4; ++kk) {
        const int kb = kk * 32 + (hi ? 8 : 0);
        f4v u0 = *(const f4v*)(Xrow + kb);
        f4v u1 = *(const f4v*)(Xrow + kb + 4);
        f4v u2 = *(const f4v*)(Xrow + kb + 16);
        f4v u3 = *(const f4v*)(Xrow + kb + 20);
        h16 t;
        #pragma unroll
        for (int i = 0; i < 4; ++i) {
            t[i]      = (_Float16)u0[i];
            t[4 + i]  = (_Float16)u1[i];
            t[8 + i]  = (_Float16)u2[i];
            t[12 + i] = (_Float16)u3[i];
        }
        ax[kk] = t;
    }

    // ---- Layer 1: H1 = silu(X W1 + b1) -> LDS ----
    {
        const _Float16* Wb = W1t + (size_t)a * IDM * DIM + (hi ? 8 : 0);
        #pragma unroll
        for (int j = 0; j < 8; ++j) {
            const int col = j * 16 + lrow;
            const _Float16* Wr = Wb + (size_t)col * DIM;
            f8 acc = {};
            #pragma unroll
            for (int kk = 0; kk < 4; ++kk) {
                h8 blo = *(const h8*)(Wr + kk * 32);
                h8 bhi = *(const h8*)(Wr + kk * 32 + 16);
                acc = wmma_f16(ax[kk], cat8(blo, bhi), acc);
            }
            const float bias = b1[a * IDM + col];
            #pragma unroll
            for (int r = 0; r < 8; ++r) {
                float v = acc[r] + bias;
                v = v / (1.0f + __expf(-v));                 // SiLU
                Hl[((hi ? 8 : 0) + r) * IDM + col] = (_Float16)v;
            }
        }
    }
    __syncthreads();

    // ---- Layer 2: H2 = silu(H1 W2 + b2) (read all A frags, then overwrite) ----
    {
        h16 ah[4];
        #pragma unroll
        for (int kk = 0; kk < 4; ++kk) {
            const _Float16* p = Hl + lrow * IDM + kk * 32 + (hi ? 8 : 0);
            ah[kk] = cat8(*(const h8*)p, *(const h8*)(p + 16));
        }
        const _Float16* Wb = W2t + (size_t)a * IDM * IDM + (hi ? 8 : 0);
        #pragma unroll
        for (int j = 0; j < 8; ++j) {
            const int col = j * 16 + lrow;
            const _Float16* Wr = Wb + (size_t)col * IDM;
            f8 acc = {};
            #pragma unroll
            for (int kk = 0; kk < 4; ++kk) {
                h8 blo = *(const h8*)(Wr + kk * 32);
                h8 bhi = *(const h8*)(Wr + kk * 32 + 16);
                acc = wmma_f16(ah[kk], cat8(blo, bhi), acc);
            }
            const float bias = b2[a * IDM + col];
            #pragma unroll
            for (int r = 0; r < 8; ++r) {
                float v = acc[r] + bias;
                v = v / (1.0f + __expf(-v));                 // SiLU
                Hl[((hi ? 8 : 0) + r) * IDM + col] = (_Float16)v;
            }
        }
    }
    __syncthreads();

    // ---- Layer 3: Y = tanh(H2 W3 + b3), FD = 32 -> 2 N-tiles ----
    {
        h16 ah[4];
        #pragma unroll
        for (int kk = 0; kk < 4; ++kk) {
            const _Float16* p = Hl + lrow * IDM + kk * 32 + (hi ? 8 : 0);
            ah[kk] = cat8(*(const h8*)p, *(const h8*)(p + 16));
        }
        const _Float16* Wb = W3t + (size_t)a * FD * IDM + (hi ? 8 : 0);
        #pragma unroll
        for (int j = 0; j < 2; ++j) {
            const int col = j * 16 + lrow;
            const _Float16* Wr = Wb + (size_t)col * IDM;
            f8 acc = {};
            #pragma unroll
            for (int kk = 0; kk < 4; ++kk) {
                h8 blo = *(const h8*)(Wr + kk * 32);
                h8 bhi = *(const h8*)(Wr + kk * 32 + 16);
                acc = wmma_f16(ah[kk], cat8(blo, bhi), acc);
            }
            const float bias = b3[a * FD + col];
            #pragma unroll
            for (int r = 0; r < 8; ++r) {
                const float v = tanhf(acc[r] + bias);
                const int row = m0 + (hi ? 8 : 0) + r;
                Y[((size_t)a * SEQ + row) * FD + col] = (_Float16)v;
            }
        }
    }
}

// ---------------------------------------------------------------------------
// Stage 2: span[a][m][n] = sigmoid( sum_k Qlift[a][m][k] * Klift[a][n][k] )
// Single WMMA per 16x16 tile (K = FD = 32). Wave reuses its A fragment over
// 4 N-tiles. Block = 8 waves -> 128 rows x 64 cols.
// grid = (SEQ/64, SEQ/128, NHEADS)
// ---------------------------------------------------------------------------
__global__ __launch_bounds__(256) void USA_span_kernel(
    const _Float16* __restrict__ Qf,   // [A, SEQ, FD]
    const _Float16* __restrict__ Kf,   // [A, SEQ, FD]
    float*          __restrict__ out)  // [A, SEQ, SEQ]
{
    const int a    = blockIdx.z;
    const int wave = threadIdx.x >> 5;
    const int lane = threadIdx.x & 31;
    const int lrow = lane & 15;
    const bool hi  = lane >= 16;
    const int m0   = blockIdx.y * 128 + wave * 16;
    const int n0b  = blockIdx.x * 64;

    // A fragment: Qlift row (m0+lrow), K chunks {0..7,16..23} / {8..15,24..31}
    const _Float16* Qrow = Qf + ((size_t)a * SEQ + m0 + lrow) * FD + (hi ? 8 : 0);
    const h16 afrag = cat8(*(const h8*)Qrow, *(const h8*)(Qrow + 16));

    #pragma unroll
    for (int j = 0; j < 4; ++j) {
        const int n0 = n0b + j * 16;
        // B fragment: Klift row (n0+lrow) = column of K^T; K {0..7,16..23} / {8..15,24..31}
        const _Float16* Krow = Kf + ((size_t)a * SEQ + n0 + lrow) * FD + (hi ? 8 : 0);
        const h16 bfrag = cat8(*(const h8*)Krow, *(const h8*)(Krow + 16));

        f8 acc = {};
        acc = wmma_f16(afrag, bfrag, acc);

        float* ob = out + ((size_t)a * SEQ + m0 + (hi ? 8 : 0)) * SEQ + n0 + lrow;
        #pragma unroll
        for (int r = 0; r < 8; ++r) {
            ob[(size_t)r * SEQ] = 1.0f / (1.0f + __expf(-acc[r]));  // sigmoid (T_ANN=0)
        }
    }
}

// ---------------------------------------------------------------------------
// Host launcher
// ---------------------------------------------------------------------------
extern "C" void kernel_launch(void* const* d_in, const int* in_sizes, int n_in,
                              void* d_out, int out_size, void* d_ws, size_t ws_size,
                              hipStream_t stream) {
    (void)in_sizes; (void)n_in; (void)out_size; (void)ws_size;

    const float* Kin = (const float*)d_in[0];
    const float* Qin = (const float*)d_in[1];
    const float* WK1 = (const float*)d_in[2];
    const float* bK1 = (const float*)d_in[3];
    const float* WK2 = (const float*)d_in[4];
    const float* bK2 = (const float*)d_in[5];
    const float* WK3 = (const float*)d_in[6];
    const float* bK3 = (const float*)d_in[7];
    const float* WQ1 = (const float*)d_in[8];
    const float* bQ1 = (const float*)d_in[9];
    const float* WQ2 = (const float*)d_in[10];
    const float* bQ2 = (const float*)d_in[11];
    const float* WQ3 = (const float*)d_in[12];
    const float* bQ3 = (const float*)d_in[13];
    float* out = (float*)d_out;

    // Workspace layout (f16 elements)
    _Float16* w = (_Float16*)d_ws;
    const size_t szW12 = (size_t)NHEADS * IDM * DIM;   // 524288
    const size_t szW3  = (size_t)NHEADS * FD * IDM;    // 131072
    const size_t szLif = (size_t)NHEADS * SEQ * FD;    // 2097152
    _Float16* WK1t = w;              w += szW12;
    _Float16* WK2t = w;              w += szW12;
    _Float16* WK3t = w;              w += szW3;
    _Float16* WQ1t = w;              w += szW12;
    _Float16* WQ2t = w;              w += szW12;
    _Float16* WQ3t = w;              w += szW3;
    _Float16* Klift = w;             w += szLif;
    _Float16* Qlift = w;             // += szLif

    // --- prep: weight transpose + f16 convert ---
    {
        const int tb = 256;
        long t12 = (long)szW12, t3 = (long)szW3;
        int g12 = (int)((t12 + tb - 1) / tb);
        int g3  = (int)((t3  + tb - 1) / tb);
        USA_transpose_f32_to_h16<<<g12, tb, 0, stream>>>(WK1, WK1t, DIM, IDM, t12);
        USA_transpose_f32_to_h16<<<g12, tb, 0, stream>>>(WK2, WK2t, IDM, IDM, t12);
        USA_transpose_f32_to_h16<<<g3,  tb, 0, stream>>>(WK3, WK3t, IDM, FD,  t3);
        USA_transpose_f32_to_h16<<<g12, tb, 0, stream>>>(WQ1, WQ1t, DIM, IDM, t12);
        USA_transpose_f32_to_h16<<<g12, tb, 0, stream>>>(WQ2, WQ2t, IDM, IDM, t12);
        USA_transpose_f32_to_h16<<<g3,  tb, 0, stream>>>(WQ3, WQ3t, IDM, FD,  t3);
    }

    // --- stage 1: uplift MLPs (WMMA) ---
    {
        dim3 grid(SEQ / 128, NHEADS);
        USA_uplift_kernel<<<grid, 256, 0, stream>>>(Kin, WK1t, bK1, WK2t, bK2,
                                                    WK3t, bK3, Klift);
        USA_uplift_kernel<<<grid, 256, 0, stream>>>(Qin, WQ1t, bQ1, WQ2t, bQ2,
                                                    WQ3t, bQ3, Qlift);
    }

    // --- stage 2: span + sigmoid (WMMA) ---
    {
        dim3 grid(SEQ / 64, SEQ / 128, NHEADS);
        USA_span_kernel<<<grid, 256, 0, stream>>>(Qlift, Klift, out);
    }
}
